// TestImageModel_47837345743274
// MI455X (gfx1250) — compile-verified, matrix operand order checked by reading, not run on hardware
//
#include <hip/hip_runtime.h>
#include <hip/hip_bf16.h>
#include <cmath>

typedef __attribute__((ext_vector_type(16))) __bf16 v16bf;
typedef __attribute__((ext_vector_type(8)))  __bf16 v8bf;
typedef __attribute__((ext_vector_type(8)))  float  v8f;

// ---------------- problem constants (match reference) ----------------
constexpr int N_NODES = 50000;
constexpr int E_EDGES = 600000;
constexpr int CIN     = 256;
constexpr int HC      = 256;   // H * C
constexpr int HEADS   = 4;
constexpr int CH      = 64;
constexpr int K1      = 25000;
constexpr int K2      = 12500;
constexpr int NP1     = 65536; // pad for bitonic sort, layer 1
constexpr int NP2     = 32768; // pad for bitonic sort, layer 2
constexpr float NEGV  = -1e30f;

// ---------------- small utility kernels ----------------
__global__ void k_fill_f32(float* p, float v, int n) {
  int t = blockIdx.x * blockDim.x + threadIdx.x;
  if (t < n) p[t] = v;
}
__global__ void k_fill_i32(int* p, int v, int n) {
  int t = blockIdx.x * blockDim.x + threadIdx.x;
  if (t < n) p[t] = v;
}
__global__ void k_f32_to_bf16(const float* __restrict__ in, __bf16* __restrict__ out, int n) {
  int t = blockIdx.x * blockDim.x + threadIdx.x;
  if (t < n) out[t] = (__bf16)in[t];
}
// W [K][Ncol] f32 -> Wt [Ncol][K] bf16
__global__ void k_transpose_to_bf16(const float* __restrict__ W, __bf16* __restrict__ Wt,
                                    int K, int Ncol) {
  int t = blockIdx.x * blockDim.x + threadIdx.x;
  if (t >= K * Ncol) return;
  int k = t / Ncol, n = t - k * Ncol;
  Wt[(size_t)n * K + k] = (__bf16)W[t];
}

// ---------------- WMMA GEMM: D[M,Ncol] = A[M,K](bf16) x Bt[Ncol,K](bf16) ----------------
// One wave computes one 16x16 output tile using v_wmma_f32_16x16x32_bf16.
__global__ __launch_bounds__(256)
void k_gemm_bf16_wmma(const __bf16* __restrict__ A, const __bf16* __restrict__ Bt,
                      float* __restrict__ D, int M, int K, int Ncol) {
  int wave = threadIdx.x >> 5;
  int lane = threadIdx.x & 31;
  int tm = blockIdx.x * 8 + wave;      // 8 waves per block, one tile-row each
  int tn = blockIdx.y;
  int row0 = tm * 16;
  int col0 = tn * 16;
  if (row0 >= M) return;               // uniform per wave

  // 16-bit A/B fragment layout (ISA 7.12.2): lane holds row/col = lane%16,
  // K-halves: lanes 0-15 -> K base 0, lanes 16-31 -> K base 8 (then +16).
  int mr = lane & 15;
  int kb = (lane >> 4) << 3;           // 0 or 8
  int arow = row0 + mr; if (arow >= M) arow = M - 1;   // clamp loads; stores predicated
  const __bf16* ap = A  + (size_t)arow * K;
  const __bf16* bp = Bt + (size_t)(col0 + mr) * K;

  v8f acc = {};
  for (int k0 = 0; k0 < K; k0 += 32) {
    __builtin_prefetch(ap + k0 + 64, 0, 0);   // global_prefetch_b8
    v8bf a0 = *(const v8bf*)(ap + k0 + kb);
    v8bf a1 = *(const v8bf*)(ap + k0 + 16 + kb);
    v8bf b0 = *(const v8bf*)(bp + k0 + kb);
    v8bf b1 = *(const v8bf*)(bp + k0 + 16 + kb);
    v16bf a, b;
#pragma unroll
    for (int i = 0; i < 8; ++i) { a[i] = a0[i]; a[i + 8] = a1[i]; b[i] = b0[i]; b[i + 8] = b1[i]; }
    acc = __builtin_amdgcn_wmma_f32_16x16x32_bf16(
        /*neg_a=*/false, a, /*neg_b=*/false, b,
        /*c_mod=*/(short)0, acc, /*reuse_a=*/false, /*reuse_b=*/false);
  }

  // D layout: lane n = lane%16 ; rows = vgpr + (lane>=16 ? 8 : 0)
  int colW = col0 + (lane & 15);
  int rbase = row0 + ((lane >> 4) << 3);
#pragma unroll
  for (int v = 0; v < 8; ++v) {
    int r = rbase + v;
    if (r < M) D[(size_t)r * Ncol + colW] = acc[v];
  }
}

// ---------------- GAT edge pipeline ----------------
__device__ __forceinline__ void atomicMaxF(float* a, float v) {
  if (v >= 0.0f) atomicMax((int*)a, __float_as_int(v));
  else atomicMin((unsigned int*)a, (unsigned int)__float_as_int(v));
}

// al_s[n,h] = sum_c h[n,h,c]*a_src[h,c] ; al_d likewise
__global__ void k_node_alpha(const float* __restrict__ h, const float* __restrict__ asrc,
                             const float* __restrict__ adst, float* __restrict__ als,
                             float* __restrict__ ald, int n) {
  int t = blockIdx.x * blockDim.x + threadIdx.x;
  if (t >= n * HEADS) return;
  int node = t >> 2, hd = t & 3;
  const float* hv = h + (size_t)node * HC + hd * CH;
  const float* as = asrc + hd * CH;
  const float* ad = adst + hd * CH;
  float ss = 0.f, dd = 0.f;
#pragma unroll 8
  for (int c = 0; c < CH; ++c) { float v = hv[c]; ss += v * as[c]; dd += v * ad[c]; }
  als[t] = ss; ald[t] = dd;
}

// e = leaky_relu(al_s[src]+al_d[dst]) ; masked -> NEG ; atomic segment-max into mmax[dst]
__global__ void k_edge_logit(const int* __restrict__ src, const int* __restrict__ dst,
                             const unsigned char* __restrict__ msk,
                             const float* __restrict__ als, const float* __restrict__ ald,
                             float* __restrict__ ebuf, float* __restrict__ mmax,
                             int ne, int nnodes) {
  int e = blockIdx.x * blockDim.x + threadIdx.x;
  if (e >= ne + nnodes) return;
  int s, d; bool m;
  if (e < ne) { s = src[e]; d = dst[e]; m = msk ? (msk[e] != 0) : true; }
  else        { s = d = e - ne; m = true; }   // self loops
#pragma unroll
  for (int hd = 0; hd < HEADS; ++hd) {
    float v = als[s * HEADS + hd] + ald[d * HEADS + hd];
    v = v > 0.f ? v : 0.2f * v;
    float ev = m ? v : NEGV;
    ebuf[(size_t)e * HEADS + hd] = ev;
    atomicMaxF(&mmax[d * HEADS + hd], ev);
  }
}

// ex = mask ? exp(e - m[dst]) : 0 ; atomic segment-sum into den[dst]
__global__ void k_edge_exp(const int* __restrict__ dst, const unsigned char* __restrict__ msk,
                           const float* __restrict__ ebuf, const float* __restrict__ mmax,
                           float* __restrict__ den, float* __restrict__ exbuf,
                           int ne, int nnodes) {
  int e = blockIdx.x * blockDim.x + threadIdx.x;
  if (e >= ne + nnodes) return;
  int d; bool m;
  if (e < ne) { d = dst[e]; m = msk ? (msk[e] != 0) : true; }
  else        { d = e - ne; m = true; }
#pragma unroll
  for (int hd = 0; hd < HEADS; ++hd) {
    float ex = m ? expf(ebuf[(size_t)e * HEADS + hd] - mmax[d * HEADS + hd]) : 0.f;
    exbuf[(size_t)e * HEADS + hd] = ex;
    if (ex != 0.f) atomicAdd(&den[d * HEADS + hd], ex);
  }
}

// out[dst] += alpha * h[src] : one wave per edge, lane covers 8 channels
__global__ void k_edge_msg(const int* __restrict__ src, const int* __restrict__ dst,
                           const unsigned char* __restrict__ msk,
                           const float* __restrict__ exbuf, const float* __restrict__ den,
                           const float* __restrict__ h, float* __restrict__ outp,
                           int ne, int nnodes) {
  int gid = blockIdx.x * blockDim.x + threadIdx.x;
  int e = gid >> 5, lane = gid & 31;
  if (e >= ne + nnodes) return;
  int s, d; bool m;
  if (e < ne) { s = src[e]; d = dst[e]; m = msk ? (msk[e] != 0) : true; }
  else        { s = d = e - ne; m = true; }
  if (!m) return;
  int hd = lane >> 3;                               // 8 ch per lane, 64 ch per head
  float ex = exbuf[(size_t)e * HEADS + hd];
  if (ex == 0.f) return;
  float alpha = ex / den[d * HEADS + hd];
  const float4* hp = (const float4*)(h + (size_t)s * HC + lane * 8);
  float4 v0 = hp[0], v1 = hp[1];
  float* op = outp + (size_t)d * HC + lane * 8;
  atomicAdd(op + 0, alpha * v0.x); atomicAdd(op + 1, alpha * v0.y);
  atomicAdd(op + 2, alpha * v0.z); atomicAdd(op + 3, alpha * v0.w);
  atomicAdd(op + 4, alpha * v1.x); atomicAdd(op + 5, alpha * v1.y);
  atomicAdd(op + 6, alpha * v1.z); atomicAdd(op + 7, alpha * v1.w);
}

__global__ void k_add_bias(float* out, const float* __restrict__ b, int n) {
  int t = blockIdx.x * blockDim.x + threadIdx.x;
  if (t < n) out[t] += b[t & (HC - 1)];
}

// ---------------- TopK pooling ----------------
__global__ void k_vec_norm256(const float* __restrict__ p, float* out) {
  __shared__ float sm[256];
  int t = threadIdx.x;
  float v = p[t];
  sm[t] = v * v;
  __syncthreads();
  for (int s = 128; s > 0; s >>= 1) { if (t < s) sm[t] += sm[t + s]; __syncthreads(); }
  if (t == 0) out[0] = sqrtf(sm[0]);
}

// score = tanh(x.p/|p|); one wave per node; pads with -inf
__global__ void k_pool_score(const float* __restrict__ x, const float* __restrict__ p,
                             const float* __restrict__ pnorm, float* __restrict__ key,
                             int* __restrict__ idx, int n, int npad) {
  int gid = blockIdx.x * blockDim.x + threadIdx.x;
  int node = gid >> 5, lane = gid & 31;
  if (node >= npad) return;
  if (node >= n) {
    if (lane == 0) { key[node] = -__builtin_inff(); idx[node] = 0x7fffffff; }
    return;
  }
  float s = 0.f;
  for (int j = lane; j < HC; j += 32) s += x[(size_t)node * HC + j] * p[j];
#pragma unroll
  for (int o = 16; o > 0; o >>= 1) s += __shfl_xor(s, o, 32);
  if (lane == 0) { key[node] = tanhf(s / pnorm[0]); idx[node] = node; }
}

// bitonic sort step, overall descending by key (ties -> ascending idx)
__global__ void k_bitonic(float* key, int* idx, int j, int k, int n) {
  int i = blockIdx.x * blockDim.x + threadIdx.x;
  if (i >= n) return;
  int ixj = i ^ j;
  if (ixj <= i) return;
  bool dirDesc = ((i & k) == 0);
  float ki = key[i], kj = key[ixj];
  int ii = idx[i], ij = idx[ixj];
  bool iBeforeJ = (ki > kj) || (ki == kj && ii < ij);
  bool inOrder = dirDesc ? iBeforeJ : !iBeforeJ;
  if (!inOrder) { key[i] = kj; key[ixj] = ki; idx[i] = ij; idx[ixj] = ii; }
}

// x_new[i] = relu(x[perm[i]] * score[i])
__global__ void k_pool_gather(const float* __restrict__ x, const float* __restrict__ key,
                              const int* __restrict__ idx, float* __restrict__ xo, int k) {
  int t = blockIdx.x * blockDim.x + threadIdx.x;
  if (t >= k * HC) return;
  int i = t >> 8, c = t & 255;
  float v = x[(size_t)idx[i] * HC + c] * key[i];
  xo[t] = v > 0.f ? v : 0.f;
}

__global__ void k_gather_i32(const int* __restrict__ in, const int* __restrict__ idx,
                             int* __restrict__ out, int k) {
  int t = blockIdx.x * blockDim.x + threadIdx.x;
  if (t < k) out[t] = in[idx[t]];
}
__global__ void k_scatter_inv(const int* __restrict__ idx, int* __restrict__ inv, int k) {
  int t = blockIdx.x * blockDim.x + threadIdx.x;
  if (t < k) inv[idx[t]] = t;
}

__global__ void k_edge_reindex(const int* __restrict__ srcIn, const int* __restrict__ dstIn,
                               const unsigned char* __restrict__ mskIn,
                               const int* __restrict__ inv,
                               int* __restrict__ srcOut, int* __restrict__ dstOut,
                               unsigned char* __restrict__ mskOut, int e) {
  int t = blockIdx.x * blockDim.x + threadIdx.x;
  if (t >= e) return;
  int s = srcIn[t], d = dstIn[t];
  bool m = mskIn ? (mskIn[t] != 0) : true;
  int ns = inv[s], nd = inv[d];
  bool nm = m && (ns >= 0) && (nd >= 0);
  srcOut[t] = nm ? ns : 0;
  dstOut[t] = nm ? nd : 0;
  mskOut[t] = nm ? 1 : 0;
}

// ---------------- output packing: (x2, [src;dst], mask, batch2) as f32 ----------------
__global__ void k_write_out(float* __restrict__ out, const float* __restrict__ x2,
                            const int* __restrict__ s2, const int* __restrict__ d2,
                            const unsigned char* __restrict__ m2, const int* __restrict__ b2) {
  int t = blockIdx.x * blockDim.x + threadIdx.x;
  const int nx = K2 * HC;
  const int total = nx + 3 * E_EDGES + K2;
  if (t >= total) return;
  if (t < nx) { out[t] = x2[t]; return; }
  int u = t - nx;
  if (u < E_EDGES) { out[t] = (float)s2[u]; return; }
  u -= E_EDGES;
  if (u < E_EDGES) { out[t] = (float)d2[u]; return; }
  u -= E_EDGES;
  if (u < E_EDGES) { out[t] = m2[u] ? 1.f : 0.f; return; }
  u -= E_EDGES;
  out[t] = (float)b2[u];
}

// ---------------- host orchestration ----------------
static inline int cdiv(int a, int b) { return (a + b - 1) / b; }

static void run_gemm(const __bf16* A, const __bf16* Bt, float* D, int M, int K, int Ncol,
                     hipStream_t st) {
  dim3 grid(cdiv(cdiv(M, 16), 8), Ncol / 16);
  k_gemm_bf16_wmma<<<grid, 256, 0, st>>>(A, Bt, D, M, K, Ncol);
}

static void run_gat(const __bf16* xb, const __bf16* Wt, float* h,
                    const float* asrc, const float* adst, const float* bias,
                    const int* src, const int* dst, const unsigned char* msk,
                    float* als, float* ald, float* mmax, float* den,
                    float* ebuf, float* exbuf, float* out,
                    int n, int ne, int K, hipStream_t st) {
  run_gemm(xb, Wt, h, n, K, HC, st);
  k_node_alpha<<<cdiv(n * HEADS, 256), 256, 0, st>>>(h, asrc, adst, als, ald, n);
  k_fill_f32<<<cdiv(n * HEADS, 256), 256, 0, st>>>(mmax, -__builtin_inff(), n * HEADS);
  k_fill_f32<<<cdiv(n * HEADS, 256), 256, 0, st>>>(den, 0.f, n * HEADS);
  k_fill_f32<<<cdiv(n * HC, 256), 256, 0, st>>>(out, 0.f, n * HC);
  int et = ne + n;
  k_edge_logit<<<cdiv(et, 256), 256, 0, st>>>(src, dst, msk, als, ald, ebuf, mmax, ne, n);
  k_edge_exp<<<cdiv(et, 256), 256, 0, st>>>(dst, msk, ebuf, mmax, den, exbuf, ne, n);
  k_edge_msg<<<cdiv(et * 32, 256), 256, 0, st>>>(src, dst, msk, exbuf, den, h, out, ne, n);
  k_add_bias<<<cdiv(n * HC, 256), 256, 0, st>>>(out, bias, n * HC);
}

static void run_pool(const float* xin, const float* p, float* pn,
                     float* key, int* idx, float* xout, int* inv,
                     const int* srcIn, const int* dstIn, const unsigned char* mskIn,
                     int* srcOut, int* dstOut, unsigned char* mskOut,
                     const int* batchIn, int* batchOut,
                     int n, int npad, int k, hipStream_t st) {
  k_vec_norm256<<<1, 256, 0, st>>>(p, pn);
  k_pool_score<<<cdiv(npad * 32, 256), 256, 0, st>>>(xin, p, pn, key, idx, n, npad);
  for (int kk = 2; kk <= npad; kk <<= 1)
    for (int j = kk >> 1; j > 0; j >>= 1)
      k_bitonic<<<cdiv(npad, 256), 256, 0, st>>>(key, idx, j, kk, npad);
  k_pool_gather<<<cdiv(k * HC, 256), 256, 0, st>>>(xin, key, idx, xout, k);
  k_gather_i32<<<cdiv(k, 256), 256, 0, st>>>(batchIn, idx, batchOut, k);
  k_fill_i32<<<cdiv(n, 256), 256, 0, st>>>(inv, -1, n);
  k_scatter_inv<<<cdiv(k, 256), 256, 0, st>>>(idx, inv, k);
  k_edge_reindex<<<cdiv(E_EDGES, 256), 256, 0, st>>>(srcIn, dstIn, mskIn, inv,
                                                     srcOut, dstOut, mskOut, E_EDGES);
}

extern "C" void kernel_launch(void* const* d_in, const int* in_sizes, int n_in,
                              void* d_out, int out_size, void* d_ws, size_t ws_size,
                              hipStream_t stream) {
  const float* x     = (const float*)d_in[0];
  const int*   ei    = (const int*)d_in[1];
  const int*   batch = (const int*)d_in[2];
  const float* W1    = (const float*)d_in[3];
  const float* asrc1 = (const float*)d_in[4];
  const float* adst1 = (const float*)d_in[5];
  const float* b1    = (const float*)d_in[6];
  const float* p1    = (const float*)d_in[7];
  const float* W2    = (const float*)d_in[8];
  const float* asrc2 = (const float*)d_in[9];
  const float* adst2 = (const float*)d_in[10];
  const float* b2    = (const float*)d_in[11];
  const float* p2    = (const float*)d_in[12];
  const int* src = ei;
  const int* dst = ei + E_EDGES;

  // bump allocator over workspace
  char* base = (char*)d_ws;
  size_t off = 0;
  auto alloc = [&](size_t bytes) -> void* {
    void* p = base + off;
    off += (bytes + 255) & ~(size_t)255;
    return p;
  };

  __bf16* xb   = (__bf16*)alloc((size_t)N_NODES * CIN * 2);
  __bf16* Wt1  = (__bf16*)alloc((size_t)CIN * HC * 2);
  __bf16* Wt2  = (__bf16*)alloc((size_t)HC * HC * 2);
  float* h1    = (float*)alloc((size_t)N_NODES * HC * 4);
  float* als1  = (float*)alloc((size_t)N_NODES * HEADS * 4);
  float* ald1  = (float*)alloc((size_t)N_NODES * HEADS * 4);
  float* m1    = (float*)alloc((size_t)N_NODES * HEADS * 4);
  float* den1  = (float*)alloc((size_t)N_NODES * HEADS * 4);
  float* ebuf1 = (float*)alloc((size_t)(E_EDGES + N_NODES) * HEADS * 4);
  float* exb1  = (float*)alloc((size_t)(E_EDGES + N_NODES) * HEADS * 4);
  float* x1    = (float*)alloc((size_t)N_NODES * HC * 4);
  float* key1  = (float*)alloc((size_t)NP1 * 4);
  int*   idx1  = (int*)alloc((size_t)NP1 * 4);
  float* x1p   = (float*)alloc((size_t)K1 * HC * 4);
  int*   inv1  = (int*)alloc((size_t)N_NODES * 4);
  int*   src1  = (int*)alloc((size_t)E_EDGES * 4);
  int*   dst1  = (int*)alloc((size_t)E_EDGES * 4);
  unsigned char* msk1 = (unsigned char*)alloc((size_t)E_EDGES);
  int*   bat1  = (int*)alloc((size_t)K1 * 4);
  __bf16* x1b  = (__bf16*)alloc((size_t)K1 * HC * 2);
  float* h2    = (float*)alloc((size_t)K1 * HC * 4);
  float* als2  = (float*)alloc((size_t)K1 * HEADS * 4);
  float* ald2  = (float*)alloc((size_t)K1 * HEADS * 4);
  float* m2    = (float*)alloc((size_t)K1 * HEADS * 4);
  float* den2  = (float*)alloc((size_t)K1 * HEADS * 4);
  float* ebuf2 = (float*)alloc((size_t)(E_EDGES + K1) * HEADS * 4);
  float* exb2  = (float*)alloc((size_t)(E_EDGES + K1) * HEADS * 4);
  float* x2    = (float*)alloc((size_t)K1 * HC * 4);
  float* key2  = (float*)alloc((size_t)NP2 * 4);
  int*   idx2  = (int*)alloc((size_t)NP2 * 4);
  float* x2p   = (float*)alloc((size_t)K2 * HC * 4);
  int*   inv2  = (int*)alloc((size_t)K1 * 4);
  int*   src2  = (int*)alloc((size_t)E_EDGES * 4);
  int*   dst2  = (int*)alloc((size_t)E_EDGES * 4);
  unsigned char* msk2 = (unsigned char*)alloc((size_t)E_EDGES);
  int*   bat2  = (int*)alloc((size_t)K2 * 4);
  float* pn    = (float*)alloc(256);
  if (off > ws_size) return;  // workspace too small; nothing safe to do

  // ---- layer 1: GATConv + TopKPool ----
  k_f32_to_bf16<<<cdiv(N_NODES * CIN, 256), 256, 0, stream>>>(x, xb, N_NODES * CIN);
  k_transpose_to_bf16<<<cdiv(CIN * HC, 256), 256, 0, stream>>>(W1, Wt1, CIN, HC);
  run_gat(xb, Wt1, h1, asrc1, adst1, b1, src, dst, nullptr,
          als1, ald1, m1, den1, ebuf1, exb1, x1, N_NODES, E_EDGES, CIN, stream);
  run_pool(x1, p1, pn + 0, key1, idx1, x1p, inv1,
           src, dst, nullptr, src1, dst1, msk1, batch, bat1,
           N_NODES, NP1, K1, stream);

  // ---- layer 2: GATConv + TopKPool ----
  k_f32_to_bf16<<<cdiv(K1 * HC, 256), 256, 0, stream>>>(x1p, x1b, K1 * HC);
  k_transpose_to_bf16<<<cdiv(HC * HC, 256), 256, 0, stream>>>(W2, Wt2, HC, HC);
  run_gat(x1b, Wt2, h2, asrc2, adst2, b2, src1, dst1, msk1,
          als2, ald2, m2, den2, ebuf2, exb2, x2, K1, E_EDGES, HC, stream);
  run_pool(x2, p2, pn + 1, key2, idx2, x2p, inv2,
           src1, dst1, msk1, src2, dst2, msk2, bat1, bat2,
           K1, NP2, K2, stream);

  // ---- pack outputs ----
  int total = K2 * HC + 3 * E_EDGES + K2;
  k_write_out<<<cdiv(total, 256), 256, 0, stream>>>((float*)d_out, x2p, src2, dst2, msk2, bat2);
}
